// MultiheadAttention_37623913513277
// MI455X (gfx1250) — compile-verified, matrix operand order checked by reading, not run on hardware
//
#include <hip/hip_runtime.h>
#include <hip/hip_bf16.h>

typedef __attribute__((ext_vector_type(16))) _Float16 v16h;
typedef __attribute__((ext_vector_type(8)))  _Float16 v8h;
typedef __attribute__((ext_vector_type(8)))  float    v8f;

#define DD 1024
#define SS 2048
#define HH 16
#define HD 64

union HV { v16h v; v8h h[2]; };

// ---- WMMA helper: D = A(16x32 f16) * B(32x16 f16) + C(16x16 f32) ----
__device__ __forceinline__ v8f wmma_f16(v16h a, v16h b, v8f c) {
  return __builtin_amdgcn_wmma_f32_16x16x32_f16(
      /*neg_a=*/false, a, /*neg_b=*/false, b,
      /*c_mod=*/(short)0, c, /*reuse_a=*/false, /*reuse_b=*/false);
}

// Load a 16x32 f16 A-fragment (also used for B fragments: pass the row-major
// transposed operand).  ISA layout: lane m = lane&15, half = lane>>4;
// VGPR0..3 hold K = half*8 .. half*8+7, VGPR4..7 hold K = 16+half*8 .. +7.
__device__ __forceinline__ v16h load_frag(const _Float16* p, int rowbase,
                                          int ldk, int k0, int lane) {
  int m = lane & 15, half = lane >> 4;
  const _Float16* base = p + (size_t)(rowbase + m) * ldk + k0 + half * 8;
  HV u;
  u.h[0] = *(const v8h*)(base);
  u.h[1] = *(const v8h*)(base + 16);
  return u.v;
}

// ---- fp32 -> f16 elementwise convert ----
__global__ __launch_bounds__(256) void cvt_f32_f16(const float* __restrict__ in,
                                                   _Float16* __restrict__ out, int n) {
  int i = blockIdx.x * 256 + threadIdx.x;
  if (i < n) out[i] = (_Float16)in[i];
}

// ---- W (1024x1024 f32) -> W^T (f16), LDS tiled transpose ----
__global__ __launch_bounds__(256) void transpose_cvt(const float* __restrict__ W,
                                                     _Float16* __restrict__ Wt) {
  __shared__ float tile[32][33];
  int tx = threadIdx.x, ty = threadIdx.y;        // 32 x 8
  int bx = blockIdx.x, by = blockIdx.y;          // col tile, row tile
#pragma unroll
  for (int i = 0; i < 4; ++i)
    tile[ty + i * 8][tx] = W[(size_t)(by * 32 + ty + i * 8) * DD + bx * 32 + tx];
  __syncthreads();
#pragma unroll
  for (int i = 0; i < 4; ++i)
    Wt[(size_t)(bx * 32 + ty + i * 8) * DD + by * 32 + tx] =
        (_Float16)tile[tx][ty + i * 8];
}

// ---- WMMA GEMM, 32x64 output tile per wave (2 A-frags x 4 B-frags) ----
// C = A(f16, M x 1024) * Wt^T + bias
// mode 0: f16 out, head layout   dst[((b*H+h)*S + s)*HD + d]   (Q, K)
// mode 1: f16 out, transposed    dst[((b*H+h)*HD + d)*S + s]   (V^T)
// mode 2: f32 out, flat          dst[m*1024 + j]               (final proj)
__global__ __launch_bounds__(32) void gemm16(const _Float16* __restrict__ A,
                                             const _Float16* __restrict__ Bt,
                                             const float* __restrict__ bias,
                                             void* __restrict__ dst, int mode) {
  int tm = blockIdx.x;         // M/32 tiles
  int tn = blockIdx.y;         // N/64 tiles
  int lane = threadIdx.x;
  v8f acc[2][4] = {};
  for (int k = 0; k < DD; k += 32) {
    v16h a0 = load_frag(A, tm * 32,      DD, k, lane);
    v16h a1 = load_frag(A, tm * 32 + 16, DD, k, lane);
    v16h b0 = load_frag(Bt, tn * 64,      DD, k, lane);
    v16h b1 = load_frag(Bt, tn * 64 + 16, DD, k, lane);
    v16h b2 = load_frag(Bt, tn * 64 + 32, DD, k, lane);
    v16h b3 = load_frag(Bt, tn * 64 + 48, DD, k, lane);
    acc[0][0] = wmma_f16(a0, b0, acc[0][0]);
    acc[0][1] = wmma_f16(a0, b1, acc[0][1]);
    acc[0][2] = wmma_f16(a0, b2, acc[0][2]);
    acc[0][3] = wmma_f16(a0, b3, acc[0][3]);
    acc[1][0] = wmma_f16(a1, b0, acc[1][0]);
    acc[1][1] = wmma_f16(a1, b1, acc[1][1]);
    acc[1][2] = wmma_f16(a1, b2, acc[1][2]);
    acc[1][3] = wmma_f16(a1, b3, acc[1][3]);
  }
  int n = lane & 15, mbase = (lane >> 4) * 8;
#pragma unroll
  for (int j = 0; j < 4; ++j) {
    int jg = tn * 64 + j * 16 + n;
    float bv = bias[jg];
    int h = jg >> 6, d = jg & (HD - 1);
#pragma unroll
    for (int i = 0; i < 2; ++i) {
#pragma unroll
      for (int r = 0; r < 8; ++r) {
        float val = acc[i][j][r] + bv;
        int mg = tm * 32 + i * 16 + mbase + r;
        if (mode == 2) {
          ((float*)dst)[(size_t)mg * DD + jg] = val;
        } else {
          int b_ = mg >> 11, s = mg & (SS - 1);
          _Float16 hv = (_Float16)val;
          if (mode == 0)
            ((_Float16*)dst)[(((size_t)(b_ * HH + h)) * SS + s) * HD + d] = hv;
          else
            ((_Float16*)dst)[(((size_t)(b_ * HH + h)) * HD + d) * SS + s] = hv;
        }
      }
    }
  }
}

// ---- Flash attention: one wave handles a 16-row q tile of one (b,h) ----
// Q,K: [BH, S, HD] f16.  Vt: [BH, HD, S] f16.  ctx out: [B, S, D] f16.
__global__ __launch_bounds__(32) void flash16(const _Float16* __restrict__ Q,
                                              const _Float16* __restrict__ K,
                                              const _Float16* __restrict__ Vt,
                                              _Float16* __restrict__ ctx) {
  __shared__ __align__(16) _Float16 pbuf[16 * 32];
  int qt = blockIdx.x;      // 0..S/16-1
  int bh = blockIdx.y;      // 0..B*H-1
  int lane = threadIdx.x;
  const _Float16* Qp = Q  + (size_t)bh * SS * HD;
  const _Float16* Kp = K  + (size_t)bh * SS * HD;
  const _Float16* Vp = Vt + (size_t)bh * HD * SS;

  v16h q0 = load_frag(Qp, qt * 16, HD, 0,  lane);
  v16h q1 = load_frag(Qp, qt * 16, HD, 32, lane);

  v8f c0 = {}, c1 = {}, c2 = {}, c3 = {};
  float mrow[8], lrow[8];
#pragma unroll
  for (int r = 0; r < 8; ++r) { mrow[r] = -1e30f; lrow[r] = 0.0f; }

  int n = lane & 15, mbase = (lane >> 4) * 8;

  for (int jb = 0; jb < SS; jb += 32) {
    // scores for 32 key columns: two 16x16 f32 tiles
    v16h k0 = load_frag(Kp, jb,      HD, 0,  lane);
    v16h k1 = load_frag(Kp, jb,      HD, 32, lane);
    v16h k2 = load_frag(Kp, jb + 16, HD, 0,  lane);
    v16h k3 = load_frag(Kp, jb + 16, HD, 32, lane);
    v8f s0 = {}, s1 = {};
    s0 = wmma_f16(q0, k0, s0);  s0 = wmma_f16(q1, k1, s0);
    s1 = wmma_f16(q0, k2, s1);  s1 = wmma_f16(q1, k3, s1);

    // online softmax: row r lives at VGPR slot r across a 16-lane half
#pragma unroll
    for (int r = 0; r < 8; ++r) {
      float a0 = s0[r] * 0.125f;            // 1/sqrt(64)
      float a1 = s1[r] * 0.125f;
      float mx = fmaxf(a0, a1);
#pragma unroll
      for (int off = 1; off < 16; off <<= 1) mx = fmaxf(mx, __shfl_xor(mx, off, 16));
      float mnew  = fmaxf(mrow[r], mx);
      float scale = __expf(mrow[r] - mnew);
      float p0 = __expf(a0 - mnew), p1 = __expf(a1 - mnew);
      float rs = p0 + p1;
#pragma unroll
      for (int off = 1; off < 16; off <<= 1) rs += __shfl_xor(rs, off, 16);
      lrow[r] = lrow[r] * scale + rs;
      mrow[r] = mnew;
      c0[r] *= scale; c1[r] *= scale; c2[r] *= scale; c3[r] *= scale;
      int m = mbase + r;
      pbuf[m * 32 + n]      = (_Float16)p0;   // C layout -> LDS (row-major P)
      pbuf[m * 32 + 16 + n] = (_Float16)p1;
    }
    __syncthreads();    // single-wave workgroup: lowers to S_NOP; orders DS ops

    // re-read P as a 16x32 A fragment
    v16h pf;
    {
      int m = lane & 15, half = lane >> 4;
      HV u;
      u.h[0] = *(const v8h*)(&pbuf[m * 32 + half * 8]);
      u.h[1] = *(const v8h*)(&pbuf[m * 32 + 16 + half * 8]);
      pf = u.v;
    }
    // ctx += P(16x32) * V(32x64): B fragments from Vt rows (contiguous)
    v16h v0 = load_frag(Vp, 0,  SS, jb, lane);
    v16h v1 = load_frag(Vp, 16, SS, jb, lane);
    v16h v2 = load_frag(Vp, 32, SS, jb, lane);
    v16h v3 = load_frag(Vp, 48, SS, jb, lane);
    c0 = wmma_f16(pf, v0, c0);
    c1 = wmma_f16(pf, v1, c1);
    c2 = wmma_f16(pf, v2, c2);
    c3 = wmma_f16(pf, v3, c3);
    __syncthreads();
  }

  // normalize rows and write ctx in [B, S, D] (= [b, s, h*64+d]) f16
  int h = bh & (HH - 1), b_ = bh >> 4;
#pragma unroll
  for (int r = 0; r < 8; ++r) {
    float inv = 1.0f / lrow[r];
    int s = qt * 16 + mbase + r;
    size_t rowoff = ((size_t)b_ * SS + s) * DD + h * HD;
    ctx[rowoff + n]           = (_Float16)(c0[r] * inv);
    ctx[rowoff + 16 + n]      = (_Float16)(c1[r] * inv);
    ctx[rowoff + 32 + n]      = (_Float16)(c2[r] * inv);
    ctx[rowoff + 48 + n]      = (_Float16)(c3[r] * inv);
  }
}

extern "C" void kernel_launch(void* const* d_in, const int* in_sizes, int n_in,
                              void* d_out, int out_size, void* d_ws, size_t ws_size,
                              hipStream_t stream) {
  (void)in_sizes; (void)n_in; (void)out_size; (void)ws_size;
  const float* x  = (const float*)d_in[0];
  const float* Wq = (const float*)d_in[1];
  const float* bq = (const float*)d_in[2];
  const float* Wk = (const float*)d_in[3];
  const float* bk = (const float*)d_in[4];
  const float* Wv = (const float*)d_in[5];
  const float* bv = (const float*)d_in[6];
  const float* Wo = (const float*)d_in[7];
  const float* bo = (const float*)d_in[8];
  float* out = (float*)d_out;

  // workspace layout (f16 elements); total 48 MB
  _Float16* ws = (_Float16*)d_ws;
  const size_t NX = (size_t)2 * SS * DD;        // 4194304 (B*S rows x D)
  const size_t NW = (size_t)DD * DD;            // 1048576
  _Float16* xh   = ws;
  _Float16* wqT  = ws + NX;
  _Float16* wkT  = wqT + NW;
  _Float16* wvT  = wkT + NW;
  _Float16* woT  = wvT + NW;
  _Float16* qh   = woT + NW;                    // [BH, S, HD]
  _Float16* kh   = qh + NX;
  _Float16* vth  = kh + NX;                     // [BH, HD, S]
  _Float16* ctxh = vth + NX;                    // [B, S, D]

  // 1) convert x to f16
  cvt_f32_f16<<<(int)((NX + 255) / 256), 256, 0, stream>>>(x, xh, (int)NX);

  // 2) transpose+convert weights
  dim3 tb(32, 8), tg(32, 32);
  transpose_cvt<<<tg, tb, 0, stream>>>(Wq, wqT);
  transpose_cvt<<<tg, tb, 0, stream>>>(Wk, wkT);
  transpose_cvt<<<tg, tb, 0, stream>>>(Wv, wvT);
  transpose_cvt<<<tg, tb, 0, stream>>>(Wo, woT);

  // 3) QKV projections: M=4096, N=1024 -> 128 x 16 tiles of 32x64, 1 wave each
  dim3 gg(128, 16);
  gemm16<<<gg, 32, 0, stream>>>(xh, wqT, bq, qh,  0);
  gemm16<<<gg, 32, 0, stream>>>(xh, wkT, bk, kh,  0);
  gemm16<<<gg, 32, 0, stream>>>(xh, wvT, bv, vth, 1);

  // 4) flash attention: 128 q-tiles x 32 (b,h)
  flash16<<<dim3(SS / 16, 2 * HH), 32, 0, stream>>>(qh, kh, vth, ctxh);

  // 5) output projection -> f32 d_out
  gemm16<<<gg, 32, 0, stream>>>(ctxh, woT, bo, out, 2);
}
